// SlidingWindowAttention_41566693490966
// MI455X (gfx1250) — matrix operand order repaired and checked by reading, not compiled, not run on hardware
//
#include <hip/hip_runtime.h>
#include <hip/hip_bf16.h>

// ---------------------------------------------------------------------------
// Sliding-window (block-diagonal) MHA for MI455X (gfx1250, wave32, WMMA).
// Phases: f32->f16 convert, QKV GEMM (f16 WMMA), flash attention per
// (batch,window,head) block (f16 WMMA), output GEMM (f16 WMMA, f32 out).
// ---------------------------------------------------------------------------

typedef __attribute__((ext_vector_type(16))) _Float16 v16h;
typedef __attribute__((ext_vector_type(8)))  _Float16 v8h;
typedef __attribute__((ext_vector_type(8)))  float    v8f;
typedef __attribute__((ext_vector_type(4)))  int      v4i;

#define DEVFN __device__ __forceinline__

// ----- async global->LDS (CDNA5), with safe fallback -----------------------
#if defined(__AMDGCN__) && __has_builtin(__builtin_amdgcn_global_load_async_to_lds_b128)
#define ASYNC_LDS 1
#endif

DEVFN void g2l_b128(void* lds, const void* g) {
#ifdef ASYNC_LDS
  __builtin_amdgcn_global_load_async_to_lds_b128(
      (__attribute__((address_space(1))) v4i*)g,
      (__attribute__((address_space(3))) v4i*)lds, 0, 0);
#else
  *(v8h*)lds = *(const v8h*)g;
#endif
}

DEVFN void g2l_wait() {
#ifdef ASYNC_LDS
  asm volatile("s_wait_asynccnt 0" ::: "memory");
#endif
}

// ----- WMMA helpers --------------------------------------------------------
// A-frag 16x32: lane = row M (lane&15); K elems [0..7] at kb+half*8,
//               elems [8..15] at kb+16+half*8.
// B-frag 32x16: lane = col N (lane&15); same K order as A-frag.
DEVFN v16h frag16(const _Float16* p0, const _Float16* p1) {
  v8h lo = *(const v8h*)p0;
  v8h hi = *(const v8h*)p1;
  v16h r;
#pragma unroll
  for (int i = 0; i < 8; ++i) { r[i] = lo[i]; r[8 + i] = hi[i]; }
  return r;
}

DEVFN v8f wmma_f16(v16h a, v16h b, v8f c) {
  return __builtin_amdgcn_wmma_f32_16x16x32_f16(false, a, false, b, (short)0, c,
                                                false, false);
}

DEVFN float redmax16(float x) {  // reduce over the 16 lanes of this half
#pragma unroll
  for (int m = 8; m >= 1; m >>= 1) x = fmaxf(x, __shfl_xor(x, m, 32));
  return x;
}
DEVFN float redsum16(float x) {
#pragma unroll
  for (int m = 8; m >= 1; m >>= 1) x += __shfl_xor(x, m, 32);
  return x;
}

// ---------------------------------------------------------------------------
// f32 -> f16 convert (vectorized x4)
// ---------------------------------------------------------------------------
__global__ __launch_bounds__(256) void cvt_f16_kernel(
    const float* __restrict__ x, _Float16* __restrict__ y, int n) {
  int i = (blockIdx.x * 256 + threadIdx.x) * 4;
  if (i < n) {
    const float4 v = *reinterpret_cast<const float4*>(x + i);
    y[i + 0] = (_Float16)v.x;
    y[i + 1] = (_Float16)v.y;
    y[i + 2] = (_Float16)v.z;
    y[i + 3] = (_Float16)v.w;
  }
}

// ---------------------------------------------------------------------------
// QKV GEMM:  qkv[row,e] = sum_k x16[row,k] * w_in16[e,k] + b_in[e]
// Epilogue scatters into Q/K: [blk][pos][d], and V transposed: [blk][d][pos],
// where blk = ((b*16+nw)*8+head), 65536 elems per blk.
// ---------------------------------------------------------------------------
__global__ __launch_bounds__(256) void qkv_gemm_kernel(
    const _Float16* __restrict__ X, const _Float16* __restrict__ W,
    const float* __restrict__ bias, _Float16* __restrict__ Qo,
    _Float16* __restrict__ Ko, _Float16* __restrict__ Vo) {
  constexpr int LDA = 72;  // 64 + 8 pad (keeps 16B alignment, spreads banks)
  __shared__ _Float16 As[128 * LDA];
  __shared__ _Float16 Bs[128 * LDA];

  const int tid  = threadIdx.x;
  const int lane = tid & 31;
  const int wid  = tid >> 5;
  const int hf   = lane >> 4;
  const int l16  = lane & 15;
  const int wm   = wid >> 2;  // 0..1
  const int wn   = wid & 3;   // 0..3
  const int row0 = blockIdx.x * 128;
  const int col0 = blockIdx.y * 128;

  v8f acc[4][2];
#pragma unroll
  for (int i = 0; i < 4; ++i)
#pragma unroll
    for (int j = 0; j < 2; ++j) acc[i][j] = (v8f)0.0f;

  for (int k0 = 0; k0 < 1024; k0 += 64) {
    __syncthreads();
#pragma unroll
    for (int j = 0; j < 4; ++j) {
      const int idx = j * 256 + tid;  // 0..1023
      const int r   = idx >> 3;       // 0..127
      const int c   = (idx & 7) * 8;  // 0..56
      g2l_b128(&As[r * LDA + c], &X[(size_t)(row0 + r) * 1024 + k0 + c]);
      g2l_b128(&Bs[r * LDA + c], &W[(size_t)(col0 + r) * 1024 + k0 + c]);
    }
    if (k0 + 64 < 1024) {
      __builtin_prefetch(&X[(size_t)(row0 + (tid >> 1)) * 1024 + k0 + 64 + (tid & 1) * 32], 0, 1);
      __builtin_prefetch(&W[(size_t)(col0 + (tid >> 1)) * 1024 + k0 + 64 + (tid & 1) * 32], 0, 1);
    }
    g2l_wait();
    __syncthreads();

#pragma unroll
    for (int ks = 0; ks < 2; ++ks) {
      const int kb = ks * 32;
      v16h af[4];
#pragma unroll
      for (int mf = 0; mf < 4; ++mf) {
        const _Float16* p = &As[(wm * 64 + mf * 16 + l16) * LDA + kb + hf * 8];
        af[mf] = frag16(p, p + 16);
      }
      v16h bf[2];
#pragma unroll
      for (int nf = 0; nf < 2; ++nf) {
        const _Float16* p = &Bs[(wn * 32 + nf * 16 + l16) * LDA + kb + hf * 8];
        bf[nf] = frag16(p, p + 16);
      }
#pragma unroll
      for (int mf = 0; mf < 4; ++mf)
#pragma unroll
        for (int nf = 0; nf < 2; ++nf)
          acc[mf][nf] = wmma_f16(af[mf], bf[nf], acc[mf][nf]);
    }
  }

  // Epilogue: bias + scatter to Q/K/V layouts.
#pragma unroll
  for (int mf = 0; mf < 4; ++mf) {
#pragma unroll
    for (int nf = 0; nf < 2; ++nf) {
      const int e    = col0 + wn * 32 + nf * 16 + l16;  // output channel
      const float bv = bias[e];
      const int part = e >> 10;       // 0=Q 1=K 2=V
      const int head = (e >> 7) & 7;
      const int d    = e & 127;
      const int rowg = row0 + wm * 64 + mf * 16 + hf * 8;  // rows rowg..rowg+7
      const int b    = rowg >> 13;
      const int s0   = rowg & 8191;
      const int nw   = s0 >> 9;
      const int pos0 = s0 & 511;
      const size_t blkbase = (size_t)((b * 16 + nw) * 8 + head) << 16;
      const v8f c = acc[mf][nf];
      if (part == 2) {
        _Float16* dst = Vo + blkbase + (size_t)d * 512 + pos0;
#pragma unroll
        for (int v = 0; v < 8; ++v) dst[v] = (_Float16)(c[v] + bv);
      } else {
        _Float16* dst = (part == 0 ? Qo : Ko) + blkbase + (size_t)pos0 * 128 + d;
#pragma unroll
        for (int v = 0; v < 8; ++v) dst[(size_t)v * 128] = (_Float16)(c[v] + bv);
      }
    }
  }
}

// ---------------------------------------------------------------------------
// Flash attention within a 512-wide window for one (b,nw,head) block.
// grid.x = 512 blocks, grid.y = 4 query tiles of 128 rows. 8 waves, each
// owning 16 query rows. K/V staged async to LDS; P converted via LDS
// (reusing the K tile) from C-frag to A-frag layout.
// ---------------------------------------------------------------------------
__global__ __launch_bounds__(256) void attn_kernel(
    const _Float16* __restrict__ Q, const _Float16* __restrict__ K,
    const _Float16* __restrict__ V, _Float16* __restrict__ O) {
  constexpr int LDK = 136;  // 128 + 8 pad
  __shared__ _Float16 Ks[128 * LDK];  // also reused as P staging
  __shared__ _Float16 Vs[128 * LDK];

  const int tid  = threadIdx.x;
  const int lane = tid & 31;
  const int wid  = tid >> 5;
  const int hf   = lane >> 4;
  const int l16  = lane & 15;
  const int blk  = blockIdx.x;  // (b*16+nw)*8+head
  const int qt   = blockIdx.y;  // 0..3

  const _Float16* Qb = Q + ((size_t)blk << 16) + (size_t)qt * 128 * 128;
  const _Float16* Kb = K + ((size_t)blk << 16);
  const _Float16* Vb = V + ((size_t)blk << 16);  // [d][pos] transposed

  // Q A-fragments stay in registers for the whole kv loop.
  v16h qf[4];
  {
    const _Float16* qrow = Qb + (size_t)(wid * 16 + l16) * 128;
#pragma unroll
    for (int ks = 0; ks < 4; ++ks)
      qf[ks] = frag16(qrow + ks * 32 + hf * 8, qrow + ks * 32 + 16 + hf * 8);
  }

  v8f of[8];
#pragma unroll
  for (int nf = 0; nf < 8; ++nf) of[nf] = (v8f)0.0f;
  float mrow[8], lrow[8];
#pragma unroll
  for (int v = 0; v < 8; ++v) { mrow[v] = -__builtin_inff(); lrow[v] = 0.0f; }

  const float SC = 0.08838834764831845f * 1.4426950408889634f;  // scale*log2e

  for (int t = 0; t < 4; ++t) {
    __syncthreads();  // everyone done with previous Ks(P)/Vs
#pragma unroll
    for (int j = 0; j < 8; ++j) {
      const int idx = j * 256 + tid;  // 0..2047
      const int r   = idx >> 4;       // 0..127
      const int c   = (idx & 15) * 8; // 0..120
      g2l_b128(&Ks[r * LDK + c], Kb + (size_t)(t * 128 + r) * 128 + c);
      g2l_b128(&Vs[r * LDK + c], Vb + (size_t)r * 512 + t * 128 + c);
    }
    g2l_wait();
    __syncthreads();

    // S = Q @ K^T  (16 q-rows x 128 kv per wave)
    v8f sf[8];
#pragma unroll
    for (int nf = 0; nf < 8; ++nf) sf[nf] = (v8f)0.0f;
#pragma unroll
    for (int ks = 0; ks < 4; ++ks) {
      const int kb = ks * 32;
#pragma unroll
      for (int nf = 0; nf < 8; ++nf) {
        const _Float16* kr = &Ks[(nf * 16 + l16) * LDK + kb + hf * 8];
        sf[nf] = wmma_f16(qf[ks], frag16(kr, kr + 16), sf[nf]);
      }
    }

    // Online softmax update (rows = hf*8 + v).
    float mnew[8], alpha[8];
#pragma unroll
    for (int v = 0; v < 8; ++v) {
      float t0 = sf[0][v];
#pragma unroll
      for (int nf = 1; nf < 8; ++nf) t0 = fmaxf(t0, sf[nf][v]);
      mnew[v]  = fmaxf(mrow[v], redmax16(t0));
      alpha[v] = exp2f((mrow[v] - mnew[v]) * SC);
    }
#pragma unroll
    for (int nf = 0; nf < 8; ++nf)
#pragma unroll
      for (int v = 0; v < 8; ++v)
        sf[nf][v] = exp2f((sf[nf][v] - mnew[v]) * SC);
#pragma unroll
    for (int v = 0; v < 8; ++v) {
      float t0 = sf[0][v];
#pragma unroll
      for (int nf = 1; nf < 8; ++nf) t0 += sf[nf][v];
      lrow[v] = lrow[v] * alpha[v] + redsum16(t0);
      mrow[v] = mnew[v];
    }
#pragma unroll
    for (int nf = 0; nf < 8; ++nf)
#pragma unroll
      for (int v = 0; v < 8; ++v) of[nf][v] = of[nf][v] * alpha[v];

    __syncthreads();  // all waves finished reading Ks before P overwrite

    // P (C-frag layout) -> LDS (wave-private 16 rows of Ks region).
#pragma unroll
    for (int nf = 0; nf < 8; ++nf)
#pragma unroll
      for (int v = 0; v < 8; ++v)
        Ks[(wid * 16 + hf * 8 + v) * LDK + nf * 16 + l16] = (_Float16)sf[nf][v];

    // O += P @ V
    const _Float16* pr = &Ks[(wid * 16 + l16) * LDK];
#pragma unroll
    for (int ks = 0; ks < 4; ++ks) {
      const int kb = ks * 32;
      const v16h pa = frag16(pr + kb + hf * 8, pr + kb + 16 + hf * 8);
#pragma unroll
      for (int nf = 0; nf < 8; ++nf) {
        const _Float16* vr = &Vs[(nf * 16 + l16) * LDK + kb + hf * 8];
        of[nf] = wmma_f16(pa, frag16(vr, vr + 16), of[nf]);
      }
    }
  }

  // Epilogue: divide by row sums, write o16 in [b][s][head*128+d] layout.
  const int b    = blk >> 7;
  const int nw   = (blk >> 3) & 15;
  const int head = blk & 7;
  const int s0   = nw * 512 + qt * 128 + wid * 16 + hf * 8;
  const size_t obase = ((size_t)b * 8192 + s0) * 1024 + head * 128;
  float inv[8];
#pragma unroll
  for (int v = 0; v < 8; ++v) inv[v] = 1.0f / lrow[v];
#pragma unroll
  for (int nf = 0; nf < 8; ++nf) {
    const int d = nf * 16 + l16;
#pragma unroll
    for (int v = 0; v < 8; ++v)
      O[obase + (size_t)v * 1024 + d] = (_Float16)(of[nf][v] * inv[v]);
  }
}

// ---------------------------------------------------------------------------
// Output GEMM: out[row,n] = sum_k o16[row,k] * w_out16[n,k] + b_out[n] (f32)
// ---------------------------------------------------------------------------
__global__ __launch_bounds__(256) void out_gemm_kernel(
    const _Float16* __restrict__ X, const _Float16* __restrict__ W,
    const float* __restrict__ bias, float* __restrict__ Out) {
  constexpr int LDA = 72;
  __shared__ _Float16 As[128 * LDA];
  __shared__ _Float16 Bs[128 * LDA];

  const int tid  = threadIdx.x;
  const int lane = tid & 31;
  const int wid  = tid >> 5;
  const int hf   = lane >> 4;
  const int l16  = lane & 15;
  const int wm   = wid >> 2;
  const int wn   = wid & 3;
  const int row0 = blockIdx.x * 128;
  const int col0 = blockIdx.y * 128;

  v8f acc[4][2];
#pragma unroll
  for (int i = 0; i < 4; ++i)
#pragma unroll
    for (int j = 0; j < 2; ++j) acc[i][j] = (v8f)0.0f;

  for (int k0 = 0; k0 < 1024; k0 += 64) {
    __syncthreads();
#pragma unroll
    for (int j = 0; j < 4; ++j) {
      const int idx = j * 256 + tid;
      const int r   = idx >> 3;
      const int c   = (idx & 7) * 8;
      g2l_b128(&As[r * LDA + c], &X[(size_t)(row0 + r) * 1024 + k0 + c]);
      g2l_b128(&Bs[r * LDA + c], &W[(size_t)(col0 + r) * 1024 + k0 + c]);
    }
    if (k0 + 64 < 1024) {
      __builtin_prefetch(&X[(size_t)(row0 + (tid >> 1)) * 1024 + k0 + 64 + (tid & 1) * 32], 0, 1);
      __builtin_prefetch(&W[(size_t)(col0 + (tid >> 1)) * 1024 + k0 + 64 + (tid & 1) * 32], 0, 1);
    }
    g2l_wait();
    __syncthreads();

#pragma unroll
    for (int ks = 0; ks < 2; ++ks) {
      const int kb = ks * 32;
      v16h af[4];
#pragma unroll
      for (int mf = 0; mf < 4; ++mf) {
        const _Float16* p = &As[(wm * 64 + mf * 16 + l16) * LDA + kb + hf * 8];
        af[mf] = frag16(p, p + 16);
      }
      v16h bf[2];
#pragma unroll
      for (int nf = 0; nf < 2; ++nf) {
        const _Float16* p = &Bs[(wn * 32 + nf * 16 + l16) * LDA + kb + hf * 8];
        bf[nf] = frag16(p, p + 16);
      }
#pragma unroll
      for (int mf = 0; mf < 4; ++mf)
#pragma unroll
        for (int nf = 0; nf < 2; ++nf)
          acc[mf][nf] = wmma_f16(af[mf], bf[nf], acc[mf][nf]);
    }
  }

#pragma unroll
  for (int mf = 0; mf < 4; ++mf) {
#pragma unroll
    for (int nf = 0; nf < 2; ++nf) {
      const int n    = col0 + wn * 32 + nf * 16 + l16;
      const float bv = bias[n];
      const int rowg = row0 + wm * 64 + mf * 16 + hf * 8;
      float* dst = Out + (size_t)rowg * 1024 + n;
      const v8f c = acc[mf][nf];
#pragma unroll
      for (int v = 0; v < 8; ++v) dst[(size_t)v * 1024] = c[v] + bv;
    }
  }
}

// ---------------------------------------------------------------------------
extern "C" void kernel_launch(void* const* d_in, const int* in_sizes, int n_in,
                              void* d_out, int out_size, void* d_ws,
                              size_t ws_size, hipStream_t stream) {
  (void)in_sizes; (void)n_in; (void)out_size; (void)ws_size;

  const float* x     = (const float*)d_in[0];
  const float* w_in  = (const float*)d_in[1];
  const float* b_in  = (const float*)d_in[2];
  const float* w_out = (const float*)d_in[3];
  const float* b_out = (const float*)d_in[4];
  float* out = (float*)d_out;

  constexpr size_t N_X  = (size_t)32768 * 1024;  // 33,554,432
  constexpr size_t N_WI = (size_t)3072 * 1024;
  constexpr size_t N_WO = (size_t)1024 * 1024;

  _Float16* x16  = (_Float16*)d_ws;
  _Float16* wi16 = x16 + N_X;
  _Float16* wo16 = wi16 + N_WI;
  _Float16* Qp   = wo16 + N_WO;
  _Float16* Kp   = Qp + N_X;
  _Float16* Vp   = Kp + N_X;
  _Float16* Op   = Vp + N_X;

  cvt_f16_kernel<<<N_X / 1024, 256, 0, stream>>>(x, x16, (int)N_X);
  cvt_f16_kernel<<<N_WI / 1024, 256, 0, stream>>>(w_in, wi16, (int)N_WI);
  cvt_f16_kernel<<<N_WO / 1024, 256, 0, stream>>>(w_out, wo16, (int)N_WO);

  qkv_gemm_kernel<<<dim3(256, 24), 256, 0, stream>>>(x16, wi16, b_in, Qp, Kp, Vp);
  attn_kernel<<<dim3(512, 4), 256, 0, stream>>>(Qp, Kp, Vp, Op);
  out_gemm_kernel<<<dim3(256, 8), 256, 0, stream>>>(Op, wo16, b_out, out);
}
